// MKGC_67413806678372
// MI455X (gfx1250) — matrix operand order repaired and checked by reading, not compiled, not run on hardware
//
#include <hip/hip_runtime.h>
#include <hip/hip_bf16.h>

typedef __attribute__((ext_vector_type(16))) __bf16 v16bf;
typedef __attribute__((ext_vector_type(8)))  float  v8f;

#define DIN   128
#define DOUT  128
#define NKER  4
#define NT    12      // K-steps of 32 over K=384 = [x|agg1|agg2]
#define NTILE 8       // DOUT / 16
#define BCHUNK_ELEMS (NT * NTILE * 512)          // bf16 elems per k-chunk = 49152
#define BCHUNK_BYTES (BCHUNK_ELEMS * 2)          // 98304 B = 96 KB

// ---------------------------------------------------------------------------
// Pre-swizzle weights into per-lane WMMA B-fragment order (bf16).
// B fragment (32x16): lane L element e -> K = 8*(L>>4) + (e&7) + 16*(e>>3), N = 16*n + (L&15)
// Wcat rows: [0,128)=Wroot, [128,256)=W1, [256,384)=W2  (pairs with x,agg1,agg2)
// flat index = ((k*NT + t)*NTILE + n)*512 + L*16 + e
// ---------------------------------------------------------------------------
__global__ void MKGC_pack_weights(const float* __restrict__ W1,
                                  const float* __restrict__ W2,
                                  const float* __restrict__ Wroot,
                                  __bf16* __restrict__ wswz, int total)
{
    int idx = blockIdx.x * blockDim.x + threadIdx.x;
    if (idx >= total) return;
    int e  = idx & 15;
    int L  = (idx >> 4) & 31;
    int n  = (idx >> 9) & 7;
    int kt = idx >> 12;
    int t  = kt % NT;
    int k  = kt / NT;
    int col  = n * 16 + (L & 15);
    int rowc = t * 32 + (L >> 4) * 8 + (e & 7) + (e >> 3) * 16;  // 0..383 in concat-K
    int seg  = rowc >> 7;
    int r    = rowc & 127;
    const float* W = (seg == 0) ? Wroot : (seg == 1) ? W1 : W2;
    wswz[idx] = (__bf16)W[((size_t)k * 128 + r) * 128 + col];
}

// ---------------------------------------------------------------------------
// Per-node dictionary coefficients: coeff = softmax(x @ Wd), one wave per node
// ---------------------------------------------------------------------------
__global__ void MKGC_coeff(const float* __restrict__ x,
                           const float* __restrict__ Wd,
                           float* __restrict__ coeff, int N)
{
    int node = (int)((blockIdx.x * blockDim.x + threadIdx.x) >> 5);
    int lane = threadIdx.x & 31;
    if (node >= N) return;
    float a0 = 0.f, a1 = 0.f, a2 = 0.f, a3 = 0.f;
#pragma unroll
    for (int j = 0; j < 4; ++j) {
        int f = lane + 32 * j;
        float  xv = x[(size_t)node * DIN + f];
        float4 w  = *reinterpret_cast<const float4*>(Wd + (size_t)f * 4);
        a0 += xv * w.x; a1 += xv * w.y; a2 += xv * w.z; a3 += xv * w.w;
    }
#pragma unroll
    for (int m = 16; m >= 1; m >>= 1) {
        a0 += __shfl_xor(a0, m, 32);
        a1 += __shfl_xor(a1, m, 32);
        a2 += __shfl_xor(a2, m, 32);
        a3 += __shfl_xor(a3, m, 32);
    }
    float mx = fmaxf(fmaxf(a0, a1), fmaxf(a2, a3));
    float e0 = __expf(a0 - mx), e1 = __expf(a1 - mx);
    float e2 = __expf(a2 - mx), e3 = __expf(a3 - mx);
    float inv = 1.f / (e0 + e1 + e2 + e3);
    if (lane == 0) {
        float4 c = make_float4(e0 * inv, e1 * inv, e2 * inv, e3 * inv);
        *reinterpret_cast<float4*>(coeff + (size_t)node * 4) = c;
    }
}

// ---------------------------------------------------------------------------
// CSR build, step 1: degree counts (only 2 atomics per edge).
// cnt1[dst]++ (src->dst aggregation), cnt2[src]++ (dst->src aggregation).
// ---------------------------------------------------------------------------
__global__ void MKGC_count(const int* __restrict__ ei,
                           int* __restrict__ cnt1, int* __restrict__ cnt2, int E)
{
    int e = blockIdx.x * blockDim.x + threadIdx.x;
    if (e >= E) return;
    int src = ei[e];
    int dst = ei[E + e];
    __hip_atomic_fetch_add(cnt1 + dst, 1, __ATOMIC_RELAXED, __HIP_MEMORY_SCOPE_AGENT);
    __hip_atomic_fetch_add(cnt2 + src, 1, __ATOMIC_RELAXED, __HIP_MEMORY_SCOPE_AGENT);
}

// ---------------------------------------------------------------------------
// CSR build, step 2: exclusive prefix sum, single 1024-thread block.
// Wave-level shfl scans + per-wave partials + running carry. off[N] = total.
// 49 chunk iterations for N=50000 -> microseconds.
// ---------------------------------------------------------------------------
__global__ void MKGC_scan(const int* __restrict__ cnt, int* __restrict__ off, int N)
{
    __shared__ int warpsum[32];
    __shared__ int carry;
    const int tid  = threadIdx.x;
    const int lane = tid & 31;
    const int wid  = tid >> 5;
    if (tid == 0) carry = 0;
    __syncthreads();
    for (int base = 0; base < N; base += 1024) {
        int i = base + tid;
        int v = (i < N) ? cnt[i] : 0;
        int s = v;                                   // inclusive scan in wave
#pragma unroll
        for (int d = 1; d < 32; d <<= 1) {
            int t = __shfl_up(s, d, 32);
            if (lane >= d) s += t;
        }
        if (lane == 31) warpsum[wid] = s;
        __syncthreads();
        if (wid == 0) {
            int w = warpsum[lane];
#pragma unroll
            for (int d = 1; d < 32; d <<= 1) {
                int t = __shfl_up(w, d, 32);
                if (lane >= d) w += t;
            }
            warpsum[lane] = w;                       // inclusive wave prefix
        }
        __syncthreads();
        int c = carry;
        int excl = c + ((wid > 0) ? warpsum[wid - 1] : 0) + (s - v);
        if (i < N) off[i] = excl;
        __syncthreads();
        if (tid == 0) carry = c + warpsum[31];
        __syncthreads();
    }
    if (tid == 0) off[N] = carry;
}

// ---------------------------------------------------------------------------
// CSR build, step 3: bucket fill (2 atomics per edge on fill cursors).
// idx1 bucket of node d holds all src with dst==d; idx2 bucket of node s
// holds all dst with src==s.
// ---------------------------------------------------------------------------
__global__ void MKGC_fill(const int* __restrict__ ei,
                          const int* __restrict__ off1, const int* __restrict__ off2,
                          int* __restrict__ fill1, int* __restrict__ fill2,
                          int* __restrict__ idx1, int* __restrict__ idx2, int E)
{
    int e = blockIdx.x * blockDim.x + threadIdx.x;
    if (e >= E) return;
    int src = ei[e];
    int dst = ei[E + e];
    int p1 = __hip_atomic_fetch_add(fill1 + dst, 1, __ATOMIC_RELAXED, __HIP_MEMORY_SCOPE_AGENT);
    idx1[off1[dst] + p1] = src;
    int p2 = __hip_atomic_fetch_add(fill2 + src, 1, __ATOMIC_RELAXED, __HIP_MEMORY_SCOPE_AGENT);
    idx2[off2[src] + p2] = dst;
}

// ---------------------------------------------------------------------------
// CSR gather-reduce: one wave per (node, direction); lane covers 4 features.
// agg[v] = (1/max(deg,1)) * sum_{u in bucket(v)} x[u]   -- no atomics, and the
// random 512B row gathers hit the 192MB L2 (x is only 25.6MB).
// ---------------------------------------------------------------------------
__global__ void MKGC_gather(const float* __restrict__ x,
                            const int* __restrict__ off1, const int* __restrict__ idx1,
                            const int* __restrict__ off2, const int* __restrict__ idx2,
                            float* __restrict__ agg1, float* __restrict__ agg2, int N)
{
    int w    = (int)((blockIdx.x * blockDim.x + threadIdx.x) >> 5);
    int lane = threadIdx.x & 31;
    int v    = w >> 1;
    int dir  = w & 1;
    if (v >= N) return;
    const int* off = dir ? off2 : off1;
    const int* idx = dir ? idx2 : idx1;
    float*     agg = dir ? agg2 : agg1;

    int b = off[v], eend = off[v + 1];
    float4 acc0 = make_float4(0.f, 0.f, 0.f, 0.f);
    float4 acc1 = make_float4(0.f, 0.f, 0.f, 0.f);
    int j = b;
    for (; j + 1 < eend; j += 2) {                   // 2-deep for latency
        int u0 = idx[j], u1 = idx[j + 1];
        float4 a = *reinterpret_cast<const float4*>(x + (size_t)u0 * DIN + lane * 4);
        float4 c = *reinterpret_cast<const float4*>(x + (size_t)u1 * DIN + lane * 4);
        acc0.x += a.x; acc0.y += a.y; acc0.z += a.z; acc0.w += a.w;
        acc1.x += c.x; acc1.y += c.y; acc1.z += c.z; acc1.w += c.w;
    }
    if (j < eend) {
        int u = idx[j];
        float4 a = *reinterpret_cast<const float4*>(x + (size_t)u * DIN + lane * 4);
        acc0.x += a.x; acc0.y += a.y; acc0.z += a.z; acc0.w += a.w;
    }
    float sc = 1.f / fmaxf((float)(eend - b), 1.f);
    float4 r = make_float4((acc0.x + acc1.x) * sc, (acc0.y + acc1.y) * sc,
                           (acc0.z + acc1.z) * sc, (acc0.w + acc1.w) * sc);
    *reinterpret_cast<float4*>(agg + (size_t)v * DIN + lane * 4) = r;
}

// ---------------------------------------------------------------------------
// Async-DMA one 96KB k-chunk of pre-swizzled B into LDS.
// Each thread copies 24 x 16B via global_load_async_to_lds_b128 (GVS mode).
// ASYNCcnt += 24 per wave.
// ---------------------------------------------------------------------------
__device__ __forceinline__ void MKGC_stage_b(const __bf16* __restrict__ gsrc,
                                             __bf16* ldst, int tid)
{
    unsigned lds  = (unsigned)(size_t)ldst + (unsigned)tid * 16u;  // low 32b = LDS offset
    unsigned voff = (unsigned)tid * 16u;
    for (int i = 0; i < BCHUNK_BYTES / (256 * 16); ++i) {          // 24 iters
        asm volatile("global_load_async_to_lds_b128 %0, %1, %2"
                     :: "v"(lds), "v"(voff), "s"(gsrc)
                     : "memory");
        lds  += 256 * 16;
        voff += 256 * 16;
    }
}

// ---------------------------------------------------------------------------
// Fused WMMA GEMM + epilogue.
// WG = 256 thr = 8 waves; wave w owns rows [blk*128 + 16w, +16), all 8 N-tiles.
// A panel (16x384 bf16; agg rows already normalized) lives in 12 v16bf
// register fragments, reused across 8 N-tiles x 4 kernels. B chunks are
// double-buffered in LDS via async DMA, read with ds_load_b128. Each n-tile
// reduction runs as two independent WMMA chains (even/odd t) to hide LDS
// latency behind the other chain's WMMA.
// ---------------------------------------------------------------------------
__global__ __launch_bounds__(256) void MKGC_gemm(
    const float* __restrict__ x,
    const float* __restrict__ agg1,
    const float* __restrict__ agg2,
    const float* __restrict__ coeff,
    const float* __restrict__ broot,
    const __bf16* __restrict__ wswz,
    float* __restrict__ out, int N)
{
    __shared__ __attribute__((aligned(32))) __bf16 sB[2][BCHUNK_ELEMS]; // 192 KB

    const int tid     = threadIdx.x;
    const int lane    = tid & 31;
    const int wave    = tid >> 5;
    const int half    = lane >> 4;
    const int mrow    = lane & 15;
    const int rowbase = blockIdx.x * 128 + wave * 16;

    // Kick off async stage of k=0 weights; overlaps the A-panel load below.
    MKGC_stage_b(wswz, &sB[0][0], tid);

    int arow = rowbase + mrow;
    if (arow > N - 1) arow = N - 1;                 // clamp (stores predicated)

    // ---- load A panel into registers, f32 -> bf16 ----
    v16bf A[NT];
#pragma unroll
    for (int t = 0; t < NT; ++t) {
        const int seg = t >> 2;
        const float* srcp = (seg == 0) ? x : (seg == 1) ? agg1 : agg2;
        const float* base = srcp + (size_t)arow * DIN + (t & 3) * 32 + half * 8;
        float4 f0 = *reinterpret_cast<const float4*>(base + 0);
        float4 f1 = *reinterpret_cast<const float4*>(base + 4);
        float4 f2 = *reinterpret_cast<const float4*>(base + 16);
        float4 f3 = *reinterpret_cast<const float4*>(base + 20);
        A[t][0]  = (__bf16)f0.x; A[t][1]  = (__bf16)f0.y;
        A[t][2]  = (__bf16)f0.z; A[t][3]  = (__bf16)f0.w;
        A[t][4]  = (__bf16)f1.x; A[t][5]  = (__bf16)f1.y;
        A[t][6]  = (__bf16)f1.z; A[t][7]  = (__bf16)f1.w;
        A[t][8]  = (__bf16)f2.x; A[t][9]  = (__bf16)f2.y;
        A[t][10] = (__bf16)f2.z; A[t][11] = (__bf16)f2.w;
        A[t][12] = (__bf16)f3.x; A[t][13] = (__bf16)f3.y;
        A[t][14] = (__bf16)f3.z; A[t][15] = (__bf16)f3.w;
    }

    // Output rows this lane produces (clamped for coeff loads).
    int grow[8];
#pragma unroll
    for (int r = 0; r < 8; ++r) {
        int gr = rowbase + r + half * 8;
        grow[r] = (gr > N - 1) ? (N - 1) : gr;
    }

    v8f tot[NTILE];
#pragma unroll
    for (int n = 0; n < NTILE; ++n) { v8f z = {}; tot[n] = z; }

    for (int k = 0; k < NKER; ++k) {
        // Prefetch next k-chunk into the other buffer (overlaps compute of k).
        if (k + 1 < NKER)
            MKGC_stage_b(wswz + (size_t)(k + 1) * BCHUNK_ELEMS,
                         &sB[(k + 1) & 1][0], tid);
        // Guarantee chunk k has landed (the newest 24 may remain in flight).
        if (k + 1 < NKER) asm volatile("s_wait_asynccnt 24" ::: "memory");
        else              asm volatile("s_wait_asynccnt 0"  ::: "memory");
        __syncthreads();

        const v16bf* Bl = reinterpret_cast<const v16bf*>(&sB[k & 1][0]);

        // softmax coefficient of kernel k for each produced row (L1/L2 hit)
        float cfk[8];
#pragma unroll
        for (int r = 0; r < 8; ++r)
            cfk[r] = coeff[(size_t)grow[r] * 4 + k];

#pragma unroll
        for (int n = 0; n < NTILE; ++n) {
            v8f acce = {};
            v8f acco = {};
#pragma unroll
            for (int t = 0; t < NT; t += 2) {        // two independent chains
                v16bf b0 = Bl[((t + 0) * NTILE + n) * 32 + lane];
                v16bf b1 = Bl[((t + 1) * NTILE + n) * 32 + lane];
                acce = __builtin_amdgcn_wmma_f32_16x16x32_bf16(
                           false, A[t + 0], false, b0, (short)0, acce, false, false);
                acco = __builtin_amdgcn_wmma_f32_16x16x32_bf16(
                           false, A[t + 1], false, b1, (short)0, acco, false, false);
            }
            v8f acc = acce + acco;
            const float br = broot[k * DOUT + n * 16 + mrow];
#pragma unroll
            for (int r = 0; r < 8; ++r) {
                float v = acc[r] + br;
                v = fmaxf(v, 0.f);                  // relu
                tot[n][r] += v * cfk[r];            // dictionary weighting
            }
        }
        __syncthreads();   // all waves done reading buf[k&1] before reuse
    }

#pragma unroll
    for (int n = 0; n < NTILE; ++n) {
#pragma unroll
        for (int r = 0; r < 8; ++r) {
            int gr = rowbase + r + half * 8;
            if (gr < N)
                __builtin_nontemporal_store(
                    tot[n][r], out + (size_t)gr * DOUT + n * 16 + mrow);
        }
    }
}

// ---------------------------------------------------------------------------
extern "C" void kernel_launch(void* const* d_in, const int* in_sizes, int n_in,
                              void* d_out, int out_size, void* d_ws, size_t ws_size,
                              hipStream_t stream)
{
    const float* x     = (const float*)d_in[0];
    const int*   ei    = (const int*)  d_in[1];
    const float* W1    = (const float*)d_in[2];
    const float* W2    = (const float*)d_in[3];
    const float* Wroot = (const float*)d_in[4];
    const float* broot = (const float*)d_in[5];
    const float* Wd    = (const float*)d_in[6];
    float*       out   = (float*)d_out;

    const int N = in_sizes[0] / DIN;
    const int E = in_sizes[1] / 2;
    const int totalW = NKER * BCHUNK_ELEMS;         // bf16 fragment elements

    char* ws = (char*)d_ws;
    size_t off = 0;
    auto take = [&](size_t bytes) -> void* {
        void* p = (void*)(ws + off);
        off += (bytes + 255) & ~(size_t)255;
        return p;
    };
    // zeroed region first: counts + fill cursors
    int*     cnt1  = (int*)    take((size_t)N * sizeof(int));
    int*     cnt2  = (int*)    take((size_t)N * sizeof(int));
    int*     fill1 = (int*)    take((size_t)N * sizeof(int));
    int*     fill2 = (int*)    take((size_t)N * sizeof(int));
    size_t zero_bytes = off;
    // fully-written buffers
    int*     off1  = (int*)    take((size_t)(N + 1) * sizeof(int));
    int*     off2  = (int*)    take((size_t)(N + 1) * sizeof(int));
    int*     idx1  = (int*)    take((size_t)E * sizeof(int));
    int*     idx2  = (int*)    take((size_t)E * sizeof(int));
    float*   agg1  = (float*)  take((size_t)N * DIN * sizeof(float));
    float*   agg2  = (float*)  take((size_t)N * DIN * sizeof(float));
    float*   coeff = (float*)  take((size_t)N * 4 * sizeof(float));
    __bf16*  wswz  = (__bf16*) take((size_t)totalW * sizeof(__bf16));

    hipMemsetAsync(ws, 0, zero_bytes, stream);

    MKGC_pack_weights<<<(totalW + 255) / 256, 256, 0, stream>>>(W1, W2, Wroot, wswz, totalW);
    MKGC_coeff<<<(N + 7) / 8, 256, 0, stream>>>(x, Wd, coeff, N);
    MKGC_count<<<(E + 255) / 256, 256, 0, stream>>>(ei, cnt1, cnt2, E);
    MKGC_scan<<<1, 1024, 0, stream>>>(cnt1, off1, N);
    MKGC_scan<<<1, 1024, 0, stream>>>(cnt2, off2, N);
    MKGC_fill<<<(E + 255) / 256, 256, 0, stream>>>(ei, off1, off2, fill1, fill2,
                                                   idx1, idx2, E);
    MKGC_gather<<<(2 * N + 7) / 8, 256, 0, stream>>>(x, off1, idx1, off2, idx2,
                                                     agg1, agg2, N);
    MKGC_gemm<<<(N + 127) / 128, 256, 0, stream>>>(x, agg1, agg2,
                                                   coeff, broot, wswz, out, N);
}
